// AttentionBlock_51728586113030
// MI455X (gfx1250) — compile-verified, matrix operand order checked by reading, not run on hardware
//
#include <hip/hip_runtime.h>

// ---------------------------------------------------------------------------
// CDNA5 (gfx1250) attention block:  out = concat(x, softmax_q(q k^T / sqrt(K)) v)
// B=8, T=2048, C=K=V=512.  Heavy GEMMs via v_wmma_f32_16x16x32_bf16 (wave32),
// W tiles staged by the Tensor Data Mover (double-buffered), register-blocked
// WMMA tiles for fragment reuse.
// ---------------------------------------------------------------------------

#define B_DIM   8
#define T_DIM   2048
#define C_DIM   512
#define M_ROWS  (B_DIM * T_DIM)        // 16384
#define SM_SCALE 0.04419417382415922f  // 1/sqrt(512)

typedef __attribute__((ext_vector_type(16))) __bf16    v16bf;
typedef __attribute__((ext_vector_type(8)))  float     v8f;
typedef __attribute__((ext_vector_type(4)))  float     f32x4;
typedef __attribute__((ext_vector_type(4)))  unsigned  u32x4;
typedef __attribute__((ext_vector_type(8)))  int       i32x8;
typedef __attribute__((ext_vector_type(4)))  int       i32x4;

union FragU {
  uint4  u[2];
  __bf16 h[16];
  v16bf  v;
};

#if defined(__has_builtin)
#  if __has_builtin(__builtin_amdgcn_tensor_load_to_lds)
#    define CDNA5_HAS_TDM 1
#  endif
#endif
#ifndef CDNA5_HAS_TDM
#  define CDNA5_HAS_TDM 0
#endif

__device__ __forceinline__ v8f wmma_bf16(v16bf a, v16bf b, v8f c) {
  // (neg_a, A, neg_b, B, c_mod, C, reuse_a, reuse_b)
  return __builtin_amdgcn_wmma_f32_16x16x32_bf16(false, a, false, b, (short)0, c,
                                                 false, false);
}

// A/B fragment load from a row-major bf16 matrix [rows][ld].
// Per-lane layout (ISA 7.12.2): r = lane&15, kb = (lane>>4)*8;
//   vector elems 0..7  = K(kb..kb+7), elems 8..15 = K(kb+16..kb+23).
__device__ __forceinline__ v16bf load_frag_bf16(const __bf16* base, int row,
                                                int k0, int ld, int lane) {
  const int r  = lane & 15;
  const int kb = (lane >> 4) * 8;
  const __bf16* p = base + (size_t)(row + r) * ld + (k0 + kb);
  FragU f;
  f.u[0] = *(const uint4*)(p);
  f.u[1] = *(const uint4*)(p + 16);
  return f.v;
}

// A fragment from an fp32 row-major matrix, converting to bf16 on the fly.
__device__ __forceinline__ v16bf load_frag_f32(const float* base, int row,
                                               int k0, int ld, int lane) {
  const int r  = lane & 15;
  const int kb = (lane >> 4) * 8;
  const float* p = base + (size_t)(row + r) * ld + (k0 + kb);
  float t[16];
  *(float4*)&t[0]  = *(const float4*)(p);
  *(float4*)&t[4]  = *(const float4*)(p + 4);
  *(float4*)&t[8]  = *(const float4*)(p + 16);
  *(float4*)&t[12] = *(const float4*)(p + 20);
  FragU f;
#pragma unroll
  for (int i = 0; i < 16; ++i) f.h[i] = (__bf16)t[i];
  return f.v;
}

// B fragment from an LDS-resident fp32 tile wt[32 k][32 n], row-major.
__device__ __forceinline__ v16bf load_frag_wlds(const float* wt, int ncol0,
                                                int lane) {
  const int n  = ncol0 + (lane & 15);
  const int kb = (lane >> 4) * 8;
  FragU f;
#pragma unroll
  for (int i = 0; i < 8; ++i) {
    const int K = (i < 4) ? (kb + 2 * i) : (16 + kb + 2 * (i - 4));
    f.h[2 * i]     = (__bf16)wt[K * 32 + n];
    f.h[2 * i + 1] = (__bf16)wt[(K + 1) * 32 + n];
  }
  return f.v;
}

#if CDNA5_HAS_TDM
// Issue a 2D TDM copy: tile_y rows x tile_x fp32 elements, row stride
// stride_x elements, from gptr into LDS at lds_off. D# per ISA 08 §8.3/8.4.
__device__ __forceinline__ void tdm_load_2d_f32(unsigned lds_off, const void* gptr,
                                                int tile_x, int tile_y,
                                                int stride_x) {
  const unsigned long long ga = (unsigned long long)(size_t)gptr;
  u32x4 g0;
  g0[0] = 1u;                                                 // count = 1
  g0[1] = lds_off;                                            // lds_addr
  g0[2] = (unsigned)(ga & 0xFFFFFFFFull);                     // global_addr lo
  g0[3] = (unsigned)((ga >> 32) & 0x1FFFFFFull) | (2u << 30); // hi | type=2
  i32x8 g1;
  g1[0] = (2 << 16);                                          // data_size = 4B
  g1[1] = (int)((unsigned)(stride_x & 0xFFFF) << 16);         // tensor_dim0 lo16
  g1[2] = (int)(((unsigned)stride_x >> 16) |
                ((unsigned)(stride_x & 0xFFFF) << 16));       // dim0 hi / dim1 lo
  g1[3] = (int)(((unsigned)stride_x >> 16) |
                ((unsigned)tile_x << 16));                    // dim1 hi / tile_dim0
  g1[4] = tile_y & 0xFFFF;                                    // tile_dim1
  g1[5] = stride_x;                                           // tensor_dim0_stride
  g1[6] = 0;
  g1[7] = 0;
  i32x4 z4 = 0;
#if __clang_major__ >= 23
  i32x8 z8 = 0;
  __builtin_amdgcn_tensor_load_to_lds(g0, g1, z4, z4, z8, 0);
#else
  __builtin_amdgcn_tensor_load_to_lds(g0, g1, z4, z4, 0);
#endif
}
#endif

// ---------------------------------------------------------------------------
// Kernel 0: pass x through into out[..., 0:512] (128-bit copies, NT stores).
// ---------------------------------------------------------------------------
__global__ __launch_bounds__(256) void copy_x_kernel(const float* __restrict__ x,
                                                     float* __restrict__ out) {
  const size_t g   = (size_t)blockIdx.x * 256 + threadIdx.x;  // 16384*128 vec4
  const size_t row = g >> 7;
  const int    c   = (int)(g & 127) * 4;
  const f32x4 v = *(const f32x4*)(x + row * 512 + c);
  __builtin_nontemporal_store(v, (f32x4*)(out + row * 1024 + c));
}

// ---------------------------------------------------------------------------
// Kernel 1: projections.  dst = bf16(x @ W + b).  grid = (M/128, 512/32, 3).
// 8 waves/block, each owns a 16x32 output strip (2 WMMA tiles, A reuse x2).
// The shared 32x32 fp32 W tile is staged by the TDM, double-buffered so the
// next DMA is in flight (s_wait_tensorcnt 1) while the current tile computes.
// ---------------------------------------------------------------------------
__global__ __launch_bounds__(256) void proj_kernel(
    const float* __restrict__ x,
    const float* __restrict__ Wq, const float* __restrict__ bq,
    const float* __restrict__ Wk, const float* __restrict__ bk,
    const float* __restrict__ Wv, const float* __restrict__ bv,
    __bf16* __restrict__ Qb, __bf16* __restrict__ Kb, __bf16* __restrict__ Vb) {
  __shared__ float wt[2][32 * 32];

  const int lane = threadIdx.x & 31;
  const int wave = threadIdx.x >> 5;
  const int row0 = blockIdx.x * 128 + wave * 16;
  const int n0   = blockIdx.y * 32;

  const float* W;
  const float* bias;
  __bf16*      dst;
  switch (blockIdx.z) {
    case 0:  W = Wq; bias = bq; dst = Qb; break;
    case 1:  W = Wk; bias = bk; dst = Kb; break;
    default: W = Wv; bias = bv; dst = Vb; break;
  }

#if CDNA5_HAS_TDM
  const unsigned off0 = (unsigned)(size_t)(void*)&wt[0][0];
  const unsigned off1 = (unsigned)(size_t)(void*)&wt[1][0];
  if (threadIdx.x < 32)                           // wave 0 drives the DMA
    tdm_load_2d_f32(off0, W + n0, 32, 32, C_DIM);
#endif

  v8f acc[2] = {};
  for (int k0 = 0; k0 < C_DIM; k0 += 32) {
    const int cur = (k0 >> 5) & 1;
#if CDNA5_HAS_TDM
    if (threadIdx.x < 32) {
      if (k0 + 32 < C_DIM) {                      // prefetch next W tile
        tdm_load_2d_f32(cur ? off0 : off1,
                        W + (size_t)(k0 + 32) * C_DIM + n0, 32, 32, C_DIM);
        __builtin_amdgcn_s_wait_tensorcnt(1);     // current tile landed
      } else {
        __builtin_amdgcn_s_wait_tensorcnt(0);
      }
    }
    __syncthreads();
#else
    {
      const int e  = threadIdx.x * 4;             // 1024 f32 / 256 threads
      const int kk = e >> 5, nn = e & 31;
      *(float4*)&wt[cur][kk * 32 + nn] =
          *(const float4*)&W[(size_t)(k0 + kk) * C_DIM + n0 + nn];
    }
    __syncthreads();
#endif
    const float* wc = wt[cur];
    const v16bf a  = load_frag_f32(x, row0, k0, C_DIM, lane);
    const v16bf b0 = load_frag_wlds(wc, 0, lane);
    const v16bf b1 = load_frag_wlds(wc, 16, lane);
    acc[0] = wmma_bf16(a, b0, acc[0]);
    acc[1] = wmma_bf16(a, b1, acc[1]);
    __syncthreads();                              // reads done before rewrite
  }

  const int hi = lane >> 4;
#pragma unroll
  for (int nt = 0; nt < 2; ++nt) {
    const int col = n0 + nt * 16 + (lane & 15);
    const float bv0 = bias[col];
#pragma unroll
    for (int i = 0; i < 8; ++i) {
      const int r = row0 + i + 8 * hi;
      dst[(size_t)r * C_DIM + col] = (__bf16)(acc[nt][i] + bv0);
    }
  }
}

// ---------------------------------------------------------------------------
// Kernel 2: column sums for the query-axis softmax:
//   colsum[b,s] = sum_{t >= s} exp(q[t].k[s] * scale)
// grid = (T/64 key-blocks, B).  Each wave register-blocks 2x2 WMMA tiles
// (32t x 32s): 4 WMMA per 4 fragment loads per k-step.
// ---------------------------------------------------------------------------
__global__ __launch_bounds__(256) void colsum_kernel(
    const __bf16* __restrict__ Qb, const __bf16* __restrict__ Kb,
    float* __restrict__ colsum) {
  __shared__ float ssum[64];
  if (threadIdx.x < 64) ssum[threadIdx.x] = 0.0f;
  __syncthreads();

  const int b     = blockIdx.y;
  const int s_blk = blockIdx.x * 64;
  const __bf16* Q = Qb + (size_t)b * T_DIM * C_DIM;
  const __bf16* K = Kb + (size_t)b * T_DIM * C_DIM;

  const int lane = threadIdx.x & 31;
  const int wave = threadIdx.x >> 5;
  const int tw   = wave >> 1;               // 0..3 : 32-row t strip
  const int sw   = wave & 1;                // 0..1 : 32-col s half
  const int s0   = s_blk + sw * 32;
  const int hi   = lane >> 4;

  float part[2] = {0.0f, 0.0f};             // per 16-wide s sub-tile
  for (int t0 = s_blk + tw * 32; t0 < T_DIM; t0 += 128) {
    if (t0 + 31 < s0) continue;             // wave tile fully above diagonal
    v8f l[2][2] = {};
    for (int k0 = 0; k0 < C_DIM; k0 += 32) {
      const v16bf a0 = load_frag_bf16(Q, t0, k0, C_DIM, lane);
      const v16bf a1 = load_frag_bf16(Q, t0 + 16, k0, C_DIM, lane);
      const v16bf b0 = load_frag_bf16(K, s0, k0, C_DIM, lane);
      const v16bf b1 = load_frag_bf16(K, s0 + 16, k0, C_DIM, lane);
      l[0][0] = wmma_bf16(a0, b0, l[0][0]);
      l[0][1] = wmma_bf16(a0, b1, l[0][1]);
      l[1][0] = wmma_bf16(a1, b0, l[1][0]);
      l[1][1] = wmma_bf16(a1, b1, l[1][1]);
    }
#pragma unroll
    for (int is = 0; is < 2; ++is) {
      const int col = s0 + is * 16 + (lane & 15);
#pragma unroll
      for (int it = 0; it < 2; ++it) {
#pragma unroll
        for (int i = 0; i < 8; ++i) {
          const int t = t0 + it * 16 + i + 8 * hi;
          if (col <= t) part[is] += __expf(l[it][is][i] * SM_SCALE);
        }
      }
    }
  }
#pragma unroll
  for (int is = 0; is < 2; ++is) {
    float p = part[is] + __shfl_xor(part[is], 16, 32);
    if (lane < 16) atomicAdd(&ssum[sw * 32 + is * 16 + lane], p);
  }
  __syncthreads();
  if (threadIdx.x < 64)
    colsum[(size_t)b * T_DIM + s_blk + threadIdx.x] = ssum[threadIdx.x];
}

// ---------------------------------------------------------------------------
// Kernel 3: read = (exp(qk*scale)/colsum[s]) @ V, causal.
// grid = (T/128, 512/128, B).  Block tile: 128(t) x 128(v), s-steps of 64.
// Phase A: 8 waves build P (128x64) with 2x2 register blocking; exp/mask/
// normalize; stash bf16 in LDS.  Phase B: 2t x 4v blocked P@V WMMAs with
// ds_load_b128 fragment reads from LDS-transposed V.
// ---------------------------------------------------------------------------
__global__ __launch_bounds__(256) void attend_kernel(
    const __bf16* __restrict__ Qb, const __bf16* __restrict__ Kb,
    const __bf16* __restrict__ Vb, const float* __restrict__ colsum,
    float* __restrict__ out) {
  __shared__ __bf16 Pl[128 * 64];    // P tile, row-major [t][s]      (16 KB)
  __shared__ __bf16 Vl[128 * 64];    // V tile transposed, [v][s]     (16 KB)

  const int b      = blockIdx.z;
  const int t_base = blockIdx.x * 128;
  const int v_base = blockIdx.y * 128;
  const __bf16* Q  = Qb + (size_t)b * T_DIM * C_DIM;
  const __bf16* K  = Kb + (size_t)b * T_DIM * C_DIM;
  const __bf16* V  = Vb + (size_t)b * T_DIM * C_DIM;
  const float*  cs = colsum + (size_t)b * T_DIM;
  float*        ob = out + (size_t)b * T_DIM * 1024;

  const int lane = threadIdx.x & 31;
  const int wave = threadIdx.x >> 5;
  const int tw   = wave >> 1;            // 0..3 : 32-row t strip (A and B)
  const int sw   = wave & 1;             // 0..1 : phase-A 32-col s half
  const int hi   = lane >> 4;

  v8f acc[2][4] = {};                    // 32t x 64v accumulators (phase B)
  const int s_end = t_base + 128;        // causal bound for this t block

  for (int s0 = 0; s0 < s_end; s0 += 64) {
    // ---- stage V[s0:s0+64, v_base:v_base+128] transposed into LDS ----
    {
      const int sr = threadIdx.x >> 2;          // 0..63
      const int cb = (threadIdx.x & 3) * 32;    // 0..96
      const __bf16* vp = V + (size_t)(s0 + sr) * C_DIM + v_base + cb;
      FragU f0, f1;
      f0.u[0] = *(const uint4*)(vp);
      f0.u[1] = *(const uint4*)(vp + 8);
      f1.u[0] = *(const uint4*)(vp + 16);
      f1.u[1] = *(const uint4*)(vp + 24);
#pragma unroll
      for (int j = 0; j < 16; ++j) {
        Vl[(cb + j) * 64 + sr]        = f0.h[j];
        Vl[(cb + 16 + j) * 64 + sr]   = f1.h[j];
      }
    }
    // ---- phase A: 32t x 32s of P per wave (2x2 WMMA blocking) ----
    {
      const int ta0 = t_base + tw * 32;
      const int sa0 = s0 + sw * 32;
      const int tl0 = tw * 32;           // local rows in Pl
      const int sl0 = sw * 32;           // local cols in Pl
      if (sa0 > ta0 + 31) {              // fully above causal diagonal
#pragma unroll
        for (int it = 0; it < 2; ++it)
#pragma unroll
          for (int i = 0; i < 8; ++i) {
            const int rl = tl0 + it * 16 + i + 8 * hi;
            Pl[rl * 64 + sl0 + (lane & 15)]      = (__bf16)0.0f;
            Pl[rl * 64 + sl0 + 16 + (lane & 15)] = (__bf16)0.0f;
          }
      } else {
        v8f l[2][2] = {};
        for (int k0 = 0; k0 < C_DIM; k0 += 32) {
          const v16bf a0 = load_frag_bf16(Q, ta0, k0, C_DIM, lane);
          const v16bf a1 = load_frag_bf16(Q, ta0 + 16, k0, C_DIM, lane);
          const v16bf b0 = load_frag_bf16(K, sa0, k0, C_DIM, lane);
          const v16bf b1 = load_frag_bf16(K, sa0 + 16, k0, C_DIM, lane);
          l[0][0] = wmma_bf16(a0, b0, l[0][0]);
          l[0][1] = wmma_bf16(a0, b1, l[0][1]);
          l[1][0] = wmma_bf16(a1, b0, l[1][0]);
          l[1][1] = wmma_bf16(a1, b1, l[1][1]);
        }
#pragma unroll
        for (int is = 0; is < 2; ++is) {
          const int col = sa0 + is * 16 + (lane & 15);
          const float rs = 1.0f / cs[col];
#pragma unroll
          for (int it = 0; it < 2; ++it) {
#pragma unroll
            for (int i = 0; i < 8; ++i) {
              const int t = ta0 + it * 16 + i + 8 * hi;
              const float e =
                  (col <= t) ? __expf(l[it][is][i] * SM_SCALE) * rs : 0.0f;
              Pl[(tl0 + it * 16 + i + 8 * hi) * 64 + sl0 + is * 16 +
                 (lane & 15)] = (__bf16)e;
            }
          }
        }
      }
    }
    __syncthreads();
    // ---- phase B: P(128x64) @ V(64x128); wave owns 32t x 64v ----
    const int vb0 = sw * 64;             // local v base for this wave
#pragma unroll
    for (int kk = 0; kk < 64; kk += 32) {
      const v16bf pa0 = load_frag_bf16(Pl, tw * 32, kk, 64, lane);
      const v16bf pa1 = load_frag_bf16(Pl, tw * 32 + 16, kk, 64, lane);
#pragma unroll
      for (int nt = 0; nt < 4; ++nt) {
        const v16bf vb = load_frag_bf16(Vl, vb0 + nt * 16, kk, 64, lane);
        acc[0][nt] = wmma_bf16(pa0, vb, acc[0][nt]);
        acc[1][nt] = wmma_bf16(pa1, vb, acc[1][nt]);
      }
    }
    __syncthreads();
  }

  const int colb = v_base + sw * 64;
#pragma unroll
  for (int it = 0; it < 2; ++it) {
#pragma unroll
    for (int nt = 0; nt < 4; ++nt) {
#pragma unroll
      for (int i = 0; i < 8; ++i) {
        const int r = t_base + tw * 32 + it * 16 + i + 8 * hi;
        __builtin_nontemporal_store(
            acc[it][nt][i],
            ob + (size_t)r * 1024 + 512 + colb + nt * 16 + (lane & 15));
      }
    }
  }
}

// ---------------------------------------------------------------------------
// Launcher.  Workspace layout (bytes):
//   [0, 16M)   Q  bf16 [16384][512]
//   [16M,32M)  K  bf16
//   [32M,48M)  V  bf16
//   [48M, +64K) colsum f32 [16384]
// ---------------------------------------------------------------------------
extern "C" void kernel_launch(void* const* d_in, const int* in_sizes, int n_in,
                              void* d_out, int out_size, void* d_ws,
                              size_t ws_size, hipStream_t stream) {
  const float* x  = (const float*)d_in[0];
  const float* Wq = (const float*)d_in[1];
  const float* bq = (const float*)d_in[2];
  const float* Wk = (const float*)d_in[3];
  const float* bk = (const float*)d_in[4];
  const float* Wv = (const float*)d_in[5];
  const float* bv = (const float*)d_in[6];
  float* out = (float*)d_out;

  const size_t qkv = (size_t)M_ROWS * C_DIM;           // elems per matrix
  __bf16* Qb = (__bf16*)d_ws;
  __bf16* Kb = Qb + qkv;
  __bf16* Vb = Kb + qkv;
  float* colsum = (float*)(Vb + qkv);

  // out[..., :512] = x
  copy_x_kernel<<<(M_ROWS * C_DIM / 4) / 256, 256, 0, stream>>>(x, out);
  // Q, K, V projections (bf16)
  proj_kernel<<<dim3(M_ROWS / 128, C_DIM / 32, 3), 256, 0, stream>>>(
      x, Wq, bq, Wk, bk, Wv, bv, Qb, Kb, Vb);
  // query-axis softmax denominators
  colsum_kernel<<<dim3(T_DIM / 64, B_DIM), 256, 0, stream>>>(Qb, Kb, colsum);
  // out[..., 512:] = P @ V
  attend_kernel<<<dim3(T_DIM / 128, C_DIM / 128, B_DIM), 256, 0, stream>>>(
      Qb, Kb, Vb, colsum, out);
}
